// Net_12910671692223
// MI455X (gfx1250) — matrix operand order repaired and checked by reading, not compiled, not run on hardware
//
#include <hip/hip_runtime.h>

typedef __attribute__((ext_vector_type(16))) __bf16 v16bf;
typedef __attribute__((ext_vector_type(8)))  __bf16 v8bf;
typedef __attribute__((ext_vector_type(8)))  float  v8f;

#define HH 256
#define WW 256
#define HWP 65536
#define NBATCH 2
#define HP 258          // padded H/W
#define CS 64           // channel stride of padded NHWC bf16 buffers
#define TPITCH 72       // LDS tile per-pixel pitch (halves): 144B -> conflict-free b128 reads

__device__ __forceinline__ unsigned short f2bf(float f) {
  __bf16 h = (__bf16)f;                 // native f32->bf16 (backend picks v_cvt)
  unsigned short u;
  __builtin_memcpy(&u, &h, 2);
  return u;
}
__device__ __forceinline__ float bf2f(unsigned short h) {
  return __uint_as_float(((unsigned int)h) << 16);
}
__device__ __forceinline__ float lrelu(float v) { return v > 0.0f ? v : 0.2f * v; }
__device__ __forceinline__ v8f zero8() { v8f z = {0.f,0.f,0.f,0.f,0.f,0.f,0.f,0.f}; return z; }

// CDNA5 async global->LDS copy (ASYNCcnt-tracked); builtins don't cover this, use ISA asm.
__device__ __forceinline__ void async_ld_b128(unsigned lds_byte_off, const void* g) {
  asm volatile("global_load_async_to_lds_b128 %0, %1, off"
               :: "v"(lds_byte_off), "v"((unsigned long long)(uintptr_t)g)
               : "memory");
}
__device__ __forceinline__ void wait_async0() {
  asm volatile("s_wait_asynccnt 0" ::: "memory");
}

// ---------------------------------------------------------------- utilities
__global__ void fill_u32(unsigned int* __restrict__ p, int n) {
  int i = blockIdx.x * blockDim.x + threadIdx.x;
  if (i < n) p[i] = 0u;
}

// OIHW f32 -> [O][tap(9)][64] bf16, zero-padded over input channels
__global__ void repack_conv(const float* __restrict__ src, unsigned short* __restrict__ dst,
                            int I, int total) {
  int i = blockIdx.x * blockDim.x + threadIdx.x;
  if (i >= total) return;
  int o = i / 576, r = i % 576;
  int tap = r >> 6, c = r & 63;
  float v = (c < I) ? src[((size_t)o * I + c) * 9 + tap] : 0.f;
  dst[i] = f2bf(v);
}

// (128,400) f32 -> (400,128) bf16
__global__ void repack_w2t(const float* __restrict__ w2, unsigned short* __restrict__ w2t) {
  int i = blockIdx.x * blockDim.x + threadIdx.x;
  if (i >= 400 * 128) return;
  int o = i >> 7, k = i & 127;
  w2t[i] = f2bf(w2[k * 400 + o]);
}

// ---------------------------------------------------------------- depthwise 7x7 branch
__global__ void wb_kernel(const float* __restrict__ x, const float* __restrict__ w,
                          float* __restrict__ out) {
  int i = blockIdx.x * blockDim.x + threadIdx.x;
  if (i >= NBATCH * 16 * HWP) return;
  int n = i >> 20, c = (i >> 16) & 15, p = i & 65535;
  int py = p >> 8, px = p & 255;
  const float* xc = x + ((size_t)(n * 16 + c) << 16);
  const float* wc = w + c * 49;
  float acc = 0.f;
  for (int ky = 0; ky < 7; ++ky) {
    int yy = py + ky - 3;
    if (yy < 0 || yy >= HH) continue;
    for (int kx = 0; kx < 7; ++kx) {
      int xx = px + kx - 3;
      if (xx < 0 || xx >= WW) continue;
      acc += xc[(yy << 8) + xx] * wc[ky * 7 + kx];
    }
  }
  out[i] = acc;
}

// ---------------------------------------------------------------- Pos2Weight MLP
__global__ void p2w_l1(const float* __restrict__ pm, const float* __restrict__ w1,
                       const float* __restrict__ b1, unsigned short* __restrict__ h1) {
  int i = blockIdx.x * blockDim.x + threadIdx.x;
  if (i >= HWP * 128) return;
  int p = i >> 7, j = i & 127;
  float v = pm[p * 2] * w1[j] + pm[p * 2 + 1] * w1[128 + j] + b1[j];
  h1[i] = f2bf(fmaxf(v, 0.f));
}

// M=65536 x K=128 x N=400 GEMM via WMMA bf16: lw = h1 @ w2 + b2
__global__ void __launch_bounds__(256) p2w_gemm(const unsigned short* __restrict__ h1,
                                                const unsigned short* __restrict__ w2t,
                                                const float* __restrict__ b2,
                                                unsigned short* __restrict__ lw) {
  const int lane = threadIdx.x & 31, wv = threadIdx.x >> 5;
  const int g = lane >> 4, ln = lane & 15;
  const int p0 = blockIdx.x * 128 + wv * 16;
  v16bf a[4];
#pragma unroll
  for (int kk = 0; kk < 4; ++kk) {
    const unsigned short* ap = h1 + (size_t)(p0 + ln) * 128 + kk * 32 + g * 8;
    v8bf lo = *(const v8bf*)ap;
    v8bf hi = *(const v8bf*)(ap + 16);
#pragma unroll
    for (int e = 0; e < 8; ++e) { a[kk][e] = lo[e]; a[kk][e + 8] = hi[e]; }
  }
  for (int nt = 0; nt < 25; ++nt) {
    v8f acc = zero8();
#pragma unroll
    for (int kk = 0; kk < 4; ++kk) {
      const unsigned short* bp = w2t + (size_t)(nt * 16 + ln) * 128 + kk * 32 + g * 8;
      const v8bf blo = *(const v8bf*)bp;
      const v8bf bhi = *(const v8bf*)(bp + 16);
      v16bf b;
#pragma unroll
      for (int e = 0; e < 8; ++e) { b[e] = blo[e]; b[e + 8] = bhi[e]; }
      acc = __builtin_amdgcn_wmma_f32_16x16x32_bf16(false, a[kk], false, b,
                                                    (short)0, acc, false, false);
    }
    const float bias = b2[nt * 16 + ln];
#pragma unroll
    for (int r = 0; r < 8; ++r) {
      int mm = g * 8 + r;
      lw[(size_t)(p0 + mm) * 400 + nt * 16 + ln] = f2bf(acc[r] + bias);
    }
  }
}

// ---------------------------------------------------------------- per-pixel 25x16 einsum
__global__ void einsum_lr(const float* __restrict__ y, const unsigned short* __restrict__ lw,
                          float* __restrict__ raw) {
  int i = blockIdx.x * blockDim.x + threadIdx.x;
  if (i >= NBATCH * HWP) return;
  int n = i >> 16, p = i & 65535, py = p >> 8, px = p & 255;
  const float* yn = y + ((size_t)n << 16);
  float acc[16];
#pragma unroll
  for (int o = 0; o < 16; ++o) acc[o] = 0.f;
  const unsigned short* lp0 = lw + (size_t)p * 400;
  for (int ky = 0; ky < 5; ++ky) {
    int yy = py + ky - 2;
    if (yy < 0 || yy >= HH) continue;
    for (int kx = 0; kx < 5; ++kx) {
      int xx = px + kx - 2;
      if (xx < 0 || xx >= WW) continue;
      float c = yn[(yy << 8) + xx];
      const unsigned short* lp = lp0 + (ky * 5 + kx) * 16;
#pragma unroll
      for (int o = 0; o < 16; ++o) acc[o] += c * bf2f(lp[o]);
    }
  }
  float* rp = raw + (size_t)i * 16;
#pragma unroll
  for (int o = 0; o < 16; ++o) rp[o] = lrelu(acc[o]);
}

// ---------------------------------------------------------------- SE helpers
template <int C>
__global__ void chan_reduce(const float* __restrict__ in, float* __restrict__ sums) {
  const int n = blockIdx.x >> 5;
  const int chunk = blockIdx.x & 31;
  const int PXB = HWP / 32;
  const int c = threadIdx.x % C;
  const int sub = threadIdx.x / C;
  const int TPC = 256 / C;
  float local = 0.f;
  const float* base = in + ((size_t)n * HWP + (size_t)chunk * PXB) * C + c;
  for (int i = sub; i < PXB; i += TPC) local += base[(size_t)i * C];
  atomicAdd(&sums[n * C + c], local);
}

template <int C, int HID>
__global__ void se_mlp(const float* __restrict__ sums, const float* __restrict__ w1,
                       const float* __restrict__ b1, const float* __restrict__ w2,
                       const float* __restrict__ b2, float* __restrict__ s) {
  __shared__ float mean[C];
  __shared__ float hid[HID];
  const int n = blockIdx.x, t = threadIdx.x;
  mean[t] = sums[n * C + t] * (1.0f / 65536.0f);
  __syncthreads();
  if (t < HID) {
    float h = b1[t];
    for (int c = 0; c < C; ++c) h += mean[c] * w1[c * HID + t];
    hid[t] = fmaxf(h, 0.f);
  }
  __syncthreads();
  float v = b2[t];
  for (int j = 0; j < HID; ++j) v += hid[j] * w2[j * C + t];
  s[n * C + t] = 1.0f / (1.0f + __expf(-v));
}

// front = lrelu(raw * s) -> padded NHWC bf16 (halo pre-zeroed)
__global__ void front_pad(const float* __restrict__ raw, const float* __restrict__ s,
                          unsigned short* __restrict__ buf) {
  int i = blockIdx.x * blockDim.x + threadIdx.x;
  if (i >= NBATCH * HWP * 16) return;
  int c = i & 15, p = (i >> 4) & 65535, n = i >> 20;
  int py = p >> 8, px = p & 255;
  float v = lrelu(raw[i] * s[n * 16 + c]);
  buf[(((size_t)n * HP + py + 1) * HP + px + 1) * CS + c] = f2bf(v);
}

// buf = lrelu(o*s + buf)   (CAB tail: gate + residual, in place on padded buffer)
__global__ void se_residual(const float* __restrict__ o, const float* __restrict__ s,
                            unsigned short* __restrict__ buf) {
  int i = blockIdx.x * blockDim.x + threadIdx.x;
  if (i >= NBATCH * HWP * 64) return;
  int c = i & 63, p = (i >> 6) & 65535, n = i >> 22;
  int py = p >> 8, px = p & 255;
  size_t bi = (((size_t)n * HP + py + 1) * HP + px + 1) * CS + c;
  float v = o[i] * s[n * 64 + c] + bf2f(buf[bi]);
  buf[bi] = f2bf(lrelu(v));
}

// d_out (NCHW f32, already holds wb) += brb conv output (NHWC f32)
__global__ void final_add(const float* __restrict__ o16, float* __restrict__ out) {
  int i = blockIdx.x * blockDim.x + threadIdx.x;
  if (i >= NBATCH * 16 * HWP) return;
  int n = i >> 20, c = (i >> 16) & 15, p = i & 65535;
  out[i] += o16[(((size_t)n << 16) + p) * 16 + c];
}

// ---------------------------------------------------------------- WMMA implicit-GEMM 3x3 conv
// A tile (3 rows x 130 px x 64ch) staged to LDS via async copy; each of 4 waves computes
// 32 pixels (two 16-px M tiles) so every B fragment load feeds two WMMAs.
template <int NT, int OUTMODE, bool ACT>
__global__ void __launch_bounds__(128) conv3x3_wmma(const unsigned short* __restrict__ in,
                                                    const unsigned short* __restrict__ wt,
                                                    const float* __restrict__ bias,
                                                    void* __restrict__ outp) {
  const int COUT = NT * 16;
  const int lane = threadIdx.x & 31, wv = threadIdx.x >> 5;
  const int g = lane >> 4, ln = lane & 15;
  const int t = blockIdx.x;          // 1024 = 2 batches * 256 rows * 2 x-halves
  const int n = t >> 9;
  const int rem = t & 511;
  const int y = rem >> 1;
  const int x0 = (rem & 1) * 128;

  __shared__ __align__(16) unsigned short tileA[3 * 130 * TPITCH];   // ~55KB

  // ---- stage halo tile: 3 rows x 130 px x 64ch, 16B per transfer, ASYNCcnt tracked
  {
    const unsigned lbase = (unsigned)(uintptr_t)(void*)tileA;
    const int total = 3 * 130 * 8;                 // 3120 b128 transfers
    for (int i = threadIdx.x; i < total; i += 128) {
      int row = i / (130 * 8);
      int r2  = i % (130 * 8);
      int px  = r2 >> 3;
      int sub = r2 & 7;
      const unsigned short* src = in + (((size_t)n * HP + y + row) * HP + (x0 + px)) * CS + sub * 8;
      unsigned dst = lbase + ((unsigned)(row * 130 + px) * TPITCH + sub * 8) * 2;
      async_ld_b128(dst, src);
    }
    wait_async0();
    __syncthreads();
  }

  const int x0w = wv * 32;                         // wave's 32-pixel strip within the 128
  v8f acc[2][NT];
#pragma unroll
  for (int mt = 0; mt < 2; ++mt)
#pragma unroll
    for (int i = 0; i < NT; ++i) acc[mt][i] = zero8();

  for (int tap = 0; tap < 9; ++tap) {
    const int ky = tap / 3, kx = tap % 3;
#pragma unroll
    for (int kk = 0; kk < 2; ++kk) {
      v16bf a[2];
#pragma unroll
      for (int mt = 0; mt < 2; ++mt) {
        const int lpx = x0w + mt * 16 + ln + kx;   // 0..129
        const unsigned short* ap = &tileA[(ky * 130 + lpx) * TPITCH + kk * 32 + g * 8];
        v8bf lo = *(const v8bf*)ap;                // ds_load_b128, conflict-free (144B pitch)
        v8bf hi = *(const v8bf*)(ap + 16);
#pragma unroll
        for (int e = 0; e < 8; ++e) { a[mt][e] = lo[e]; a[mt][e + 8] = hi[e]; }
      }
#pragma unroll
      for (int nt = 0; nt < NT; ++nt) {
        const unsigned short* bp = wt + ((size_t)(nt * 16 + ln) * 9 + tap) * 64 +
                                        kk * 32 + g * 8;
        const v8bf blo = *(const v8bf*)bp;
        const v8bf bhi = *(const v8bf*)(bp + 16);
        v16bf b;
#pragma unroll
        for (int e = 0; e < 8; ++e) { b[e] = blo[e]; b[e + 8] = bhi[e]; }
        acc[0][nt] = __builtin_amdgcn_wmma_f32_16x16x32_bf16(false, a[0], false, b,
                                                             (short)0, acc[0][nt], false, false);
        acc[1][nt] = __builtin_amdgcn_wmma_f32_16x16x32_bf16(false, a[1], false, b,
                                                             (short)0, acc[1][nt], false, false);
      }
    }
  }

#pragma unroll
  for (int mt = 0; mt < 2; ++mt)
#pragma unroll
    for (int nt = 0; nt < NT; ++nt) {
      const int ch = nt * 16 + ln;                 // D: N = lane&15
      const float bv = bias[ch];
#pragma unroll
      for (int r = 0; r < 8; ++r) {
        const int mm = g * 8 + r;                  // D: M = r + 8*(lane>=16)
        const int xx = x0 + x0w + mt * 16 + mm;
        float v = acc[mt][nt][r] + bv;
        if (ACT) v = lrelu(v);
        if (OUTMODE == 0) {
          ((unsigned short*)outp)[(((size_t)n * HP + y + 1) * HP + xx + 1) * CS + ch] = f2bf(v);
        } else {
          ((float*)outp)[(((size_t)n * HH + y) * WW + xx) * COUT + ch] = v;
        }
      }
    }
}

// ---------------------------------------------------------------- host launcher
extern "C" void kernel_launch(void* const* d_in, const int* in_sizes, int n_in,
                              void* d_out, int out_size, void* d_ws, size_t ws_size,
                              hipStream_t stream) {
  (void)in_sizes; (void)n_in; (void)out_size; (void)ws_size;
  const float* x       = (const float*)d_in[0];
  const float* y       = (const float*)d_in[1];
  const float* pos     = (const float*)d_in[2];
  const float* wbw     = (const float*)d_in[3];
  const float* p2w_w1  = (const float*)d_in[4];
  const float* p2w_b1  = (const float*)d_in[5];
  const float* p2w_w2  = (const float*)d_in[6];
  const float* p2w_b2  = (const float*)d_in[7];
  const float* fin_w   = (const float*)d_in[8];
  const float* fin_b   = (const float*)d_in[9];
  const float* cw[6]   = {(const float*)d_in[10], (const float*)d_in[12], (const float*)d_in[14],
                          (const float*)d_in[20], (const float*)d_in[22], (const float*)d_in[24]};
  const float* cb[6]   = {(const float*)d_in[11], (const float*)d_in[13], (const float*)d_in[15],
                          (const float*)d_in[21], (const float*)d_in[23], (const float*)d_in[25]};
  const float* f1maw1  = (const float*)d_in[16];
  const float* f1mab1  = (const float*)d_in[17];
  const float* f1maw2  = (const float*)d_in[18];
  const float* f1mab2  = (const float*)d_in[19];
  const float* f2maw1  = (const float*)d_in[26];
  const float* f2mab1  = (const float*)d_in[27];
  const float* f2maw2  = (const float*)d_in[28];
  const float* f2mab2  = (const float*)d_in[29];
  const float* brfw1   = (const float*)d_in[30];
  const float* brfb1   = (const float*)d_in[31];
  const float* brfw2   = (const float*)d_in[32];
  const float* brfb2   = (const float*)d_in[33];
  const float* brb_w   = (const float*)d_in[34];
  const float* brb_b   = (const float*)d_in[35];

  char* ws = (char*)d_ws;
  size_t off = 0;
  auto carve = [&](size_t bytes) -> void* {
    void* p = ws + off;
    off += (bytes + 255) & ~(size_t)255;
    return p;
  };
  unsigned short* LW   = (unsigned short*)carve((size_t)HWP * 400 * 2);
  unsigned short* H1   = (unsigned short*)carve((size_t)HWP * 128 * 2);
  unsigned short* W2T  = (unsigned short*)carve((size_t)400 * 128 * 2);
  unsigned short* WFIN = (unsigned short*)carve((size_t)64 * 576 * 2);
  unsigned short* WC[6];
  for (int i = 0; i < 6; ++i) WC[i] = (unsigned short*)carve((size_t)64 * 576 * 2);
  unsigned short* WBRB = (unsigned short*)carve((size_t)16 * 576 * 2);
  const size_t PADB = (size_t)NBATCH * HP * HP * CS * 2;
  unsigned short* BUF16 = (unsigned short*)carve(PADB);   // front buffer, reused as bufC
  unsigned short* BUFA  = (unsigned short*)carve(PADB);
  unsigned short* BUFB  = (unsigned short*)carve(PADB);
  float* O64   = (float*)carve((size_t)NBATCH * HWP * 64 * 4);
  float* RAW   = (float*)carve((size_t)NBATCH * HWP * 16 * 4);
  float* SUM64 = (float*)carve(2 * 64 * 4);
  float* S64   = (float*)carve(2 * 64 * 4);
  float* SUM16 = (float*)carve(2 * 16 * 4);
  float* S16   = (float*)carve(2 * 16 * 4);
  float* O16   = O64;   // brb output reuses O64 region

  const int TB = 256;
  auto cdiv = [](long a, long b) { return (int)((a + b - 1) / b); };

  const int padu32 = (int)(PADB / 4);
  fill_u32<<<cdiv(padu32, TB), TB, 0, stream>>>((unsigned int*)BUF16, padu32);
  fill_u32<<<cdiv(padu32, TB), TB, 0, stream>>>((unsigned int*)BUFA, padu32);
  fill_u32<<<cdiv(padu32, TB), TB, 0, stream>>>((unsigned int*)BUFB, padu32);

  repack_w2t<<<cdiv(400 * 128, TB), TB, 0, stream>>>(p2w_w2, W2T);
  repack_conv<<<cdiv(64 * 576, TB), TB, 0, stream>>>(fin_w, WFIN, 16, 64 * 576);
  for (int i = 0; i < 6; ++i)
    repack_conv<<<cdiv(64 * 576, TB), TB, 0, stream>>>(cw[i], WC[i], 64, 64 * 576);
  repack_conv<<<cdiv(16 * 576, TB), TB, 0, stream>>>(brb_w, WBRB, 64, 16 * 576);

  wb_kernel<<<cdiv((long)NBATCH * 16 * HWP, TB), TB, 0, stream>>>(x, wbw, (float*)d_out);

  p2w_l1<<<cdiv((long)HWP * 128, TB), TB, 0, stream>>>(pos, p2w_w1, p2w_b1, H1);
  p2w_gemm<<<512, 256, 0, stream>>>(H1, W2T, p2w_b2, LW);

  einsum_lr<<<cdiv((long)NBATCH * HWP, TB), TB, 0, stream>>>(y, LW, RAW);

  fill_u32<<<1, 64, 0, stream>>>((unsigned int*)SUM16, 32);
  chan_reduce<16><<<64, 256, 0, stream>>>(RAW, SUM16);
  se_mlp<16, 4><<<2, 16, 0, stream>>>(SUM16, brfw1, brfb1, brfw2, brfb2, S16);
  front_pad<<<cdiv((long)NBATCH * HWP * 16, TB), TB, 0, stream>>>(RAW, S16, BUF16);

  conv3x3_wmma<4, 0, false><<<1024, 128, 0, stream>>>(BUF16, WFIN, fin_b, BUFA);

  conv3x3_wmma<4, 0, true><<<1024, 128, 0, stream>>>(BUFA, WC[0], cb[0], BUFB);
  conv3x3_wmma<4, 0, true><<<1024, 128, 0, stream>>>(BUFB, WC[1], cb[1], BUF16);
  conv3x3_wmma<4, 1, false><<<1024, 128, 0, stream>>>(BUF16, WC[2], cb[2], O64);
  fill_u32<<<1, 128, 0, stream>>>((unsigned int*)SUM64, 128);
  chan_reduce<64><<<64, 256, 0, stream>>>(O64, SUM64);
  se_mlp<64, 16><<<2, 64, 0, stream>>>(SUM64, f1maw1, f1mab1, f1maw2, f1mab2, S64);
  se_residual<<<cdiv((long)NBATCH * HWP * 64, TB), TB, 0, stream>>>(O64, S64, BUFA);

  conv3x3_wmma<4, 0, true><<<1024, 128, 0, stream>>>(BUFA, WC[3], cb[3], BUFB);
  conv3x3_wmma<4, 0, true><<<1024, 128, 0, stream>>>(BUFB, WC[4], cb[4], BUF16);
  conv3x3_wmma<4, 1, false><<<1024, 128, 0, stream>>>(BUF16, WC[5], cb[5], O64);
  fill_u32<<<1, 128, 0, stream>>>((unsigned int*)SUM64, 128);
  chan_reduce<64><<<64, 256, 0, stream>>>(O64, SUM64);
  se_mlp<64, 16><<<2, 64, 0, stream>>>(SUM64, f2maw1, f2mab1, f2maw2, f2mab2, S64);
  se_residual<<<cdiv((long)NBATCH * HWP * 64, TB), TB, 0, stream>>>(O64, S64, BUFA);

  conv3x3_wmma<1, 1, false><<<1024, 128, 0, stream>>>(BUFA, WBRB, brb_b, O16);
  final_add<<<cdiv((long)NBATCH * 16 * HWP, TB), TB, 0, stream>>>(O16, (float*)d_out);
}
